// ExpertRouter_58342835749139
// MI455X (gfx1250) — matrix operand order repaired and checked by reading, not compiled, not run on hardware
//
#include <hip/hip_runtime.h>
#include <hip/hip_bf16.h>
#include <math.h>

typedef __attribute__((ext_vector_type(16))) _Float16 v16h;
typedef __attribute__((ext_vector_type(8)))  float    v8f;
typedef __attribute__((ext_vector_type(4)))  float    v4f;

#define DDIM 1024
#define EEXP 2
#define WAVES_PER_BLOCK 8
#define TOKENS_PER_WAVE 16
#define TOKENS_PER_BLOCK (WAVES_PER_BLOCK * TOKENS_PER_WAVE)   // 128

// rows of the fused weight matrix W[4][D]: 0=gate0 1=gate1 2=noise0 3=noise1
// (rows 4..15 zero-padded so WMMA B columns 4..15 contribute nothing)

__global__ __launch_bounds__(256)
void ExpertRouter_gating_kernel(const float* __restrict__ x,            // [BN, D]
                                const float* __restrict__ gate_w,      // [E, D]
                                const float* __restrict__ noise_w,     // [E, D]
                                const float* __restrict__ noise_sample,// [BN, E]
                                float* __restrict__ out_weights,       // [BN, E]
                                float* __restrict__ out_decisions,     // [BN]
                                float* __restrict__ accum)             // [3]
{
    __shared__ __align__(32) _Float16 Wlds[16 * DDIM];                 // 32 KB
    __shared__ float Cst[WAVES_PER_BLOCK][16][16];                     // 8 KB

    const int tid = threadIdx.x;

    // Stage fused (zero-padded) weight matrix into LDS as f16.
    for (int i = tid; i < 16 * DDIM; i += 256) {
        const int row = i >> 10;          // /DDIM
        const int col = i & (DDIM - 1);
        float v = 0.0f;
        if (row < 2)       v = gate_w[row * DDIM + col];
        else if (row < 4)  v = noise_w[(row - 2) * DDIM + col];
        Wlds[i] = (_Float16)v;
    }
    __syncthreads();

    const int wave = tid >> 5;
    const int lane = tid & 31;
    const int m    = lane & 15;           // row (A) / column (B)
    const int hsel = lane >> 4;           // half-wave select
    const int tokBase = blockIdx.x * TOKENS_PER_BLOCK + wave * TOKENS_PER_WAVE;

    const float* xrow = x + (size_t)(tokBase + m) * DDIM;

    v8f acc = {};
    #pragma unroll 4
    for (int k = 0; k < DDIM; k += 32) {
        // ---- A fragment: 16x32 f16 layout (lane m holds K = 16*hsel..+15)
        const v4f a0 = *(const v4f*)(xrow + k + 16 * hsel + 0);
        const v4f a1 = *(const v4f*)(xrow + k + 16 * hsel + 4);
        const v4f a2 = *(const v4f*)(xrow + k + 16 * hsel + 8);
        const v4f a3 = *(const v4f*)(xrow + k + 16 * hsel + 12);
        v16h a;
        #pragma unroll
        for (int e = 0; e < 4; ++e) {
            a[e]      = (_Float16)a0[e];
            a[4 + e]  = (_Float16)a1[e];
            a[8 + e]  = (_Float16)a2[e];
            a[12 + e] = (_Float16)a3[e];
        }
        // ---- B fragment: 32x16 f16 layout (lane = column m; element e = W[m][k + 16*hsel + e])
        const v16h b = *(const v16h*)(&Wlds[m * DDIM + k + 16 * hsel]);

        acc = __builtin_amdgcn_wmma_f32_16x16x32_f16(
                  false, a, false, b, (short)0, acc, false, false);

        if (k + 64 < DDIM)
            __builtin_prefetch(xrow + k + 64, 0, 1);
    }

    // C tile -> LDS: lane holds C[M][N] with N = m, M = j + 8*hsel for VGPR j.
    #pragma unroll
    for (int j = 0; j < 8; ++j)
        Cst[wave][j + 8 * hsel][m] = acc[j];
    __syncthreads();

    // ---- Per-token epilogue (lanes 0..15 of each wave own one token each) ----
    float ent = 0.0f, asum = 0.0f, bsum = 0.0f;
    if (lane < 16) {
        const int tok = tokBase + lane;
        const float cg0 = Cst[wave][lane][0];
        const float cg1 = Cst[wave][lane][1];
        const float cn0 = Cst[wave][lane][2];
        const float cn1 = Cst[wave][lane][3];
        const float ns0 = noise_sample[tok * 2 + 0];
        const float ns1 = noise_sample[tok * 2 + 1];

        // softplus(v) + NOISE_EPS, overflow-safe
        const float sp0 = fmaxf(cn0, 0.0f) + log1pf(expf(-fabsf(cn0))) + 0.01f;
        const float sp1 = fmaxf(cn1, 0.0f) + log1pf(expf(-fabsf(cn1))) + 0.01f;
        const float l0 = cg0 + ns0 * sp0;
        const float l1 = cg1 + ns1 * sp1;

        // 2-way softmax
        const float mx  = fmaxf(l0, l1);
        const float e0  = expf(l0 - mx);
        const float e1  = expf(l1 - mx);
        const float inv = 1.0f / (e0 + e1);
        const float p0  = e0 * inv;
        const float p1  = e1 * inv;

        ent = -(p0 * logf(p0 + 1e-10f) + p1 * logf(p1 + 1e-10f));

        // top-p with E=2: argsort(-logits) is stable -> ties pick expert 0
        const int   top  = (p1 > p0) ? 1 : 0;
        const float pmax = top ? p1 : p0;
        const float pmin = top ? p0 : p1;
        const int   only_top = (pmax > 0.7f);

        const float w_other = only_top ? 0.0f : 1.0f;
        out_weights[tok * 2 + 0] = (top == 0) ? 1.0f : w_other;
        out_weights[tok * 2 + 1] = (top == 1) ? 1.0f : w_other;
        out_decisions[tok]       = only_top ? 0.0f : 1.0f;

        asum = pmax;                               // importance[sorted pos 0]
        bsum = only_top ? 0.0f : pmin;             // importance[sorted pos 1]
    }

    // wave32 reduction (lanes 16..31 contribute zeros)
    #pragma unroll
    for (int off = 16; off > 0; off >>= 1) {
        ent  += __shfl_down(ent,  off, 32);
        asum += __shfl_down(asum, off, 32);
        bsum += __shfl_down(bsum, off, 32);
    }
    if (lane == 0) {
        atomicAdd(&accum[0], ent);
        atomicAdd(&accum[1], asum);
        atomicAdd(&accum[2], bsum);
    }
}

__global__ void ExpertRouter_finalize_kernel(const float* __restrict__ accum,
                                             float* __restrict__ loss_out,
                                             int bn)
{
    const float ent = accum[0];
    const float a   = accum[1];
    const float b   = accum[2];
    const float mean = 0.5f * (a + b);
    // unbiased variance of 2 samples: sum((x-mean)^2)/(2-1)
    const float var  = (a - mean) * (a - mean) + (b - mean) * (b - mean);
    const float loss_importance = var / (mean * mean + 1e-10f);
    const float loss_dynamic    = ent / (float)bn;
    loss_out[0] = loss_importance + 0.1f * loss_dynamic;
}

extern "C" void kernel_launch(void* const* d_in, const int* in_sizes, int n_in,
                              void* d_out, int out_size, void* d_ws, size_t ws_size,
                              hipStream_t stream) {
    const float* x            = (const float*)d_in[0];   // [B,N,D]
    const float* gate_w       = (const float*)d_in[1];   // [E,D]
    const float* noise_w      = (const float*)d_in[2];   // [E,D]
    const float* noise_sample = (const float*)d_in[3];   // [BN,E]

    const int bn = in_sizes[3] / EEXP;                   // B*N tokens

    float* out       = (float*)d_out;
    float* out_w     = out;                              // BN*E
    float* out_dec   = out + (size_t)bn * EEXP;          // BN
    float* out_loss  = out + (size_t)bn * EEXP + bn;     // 1

    float* accum = (float*)d_ws;                         // 3 floats
    hipMemsetAsync(accum, 0, 3 * sizeof(float), stream);

    const int nblocks = bn / TOKENS_PER_BLOCK;           // 256 for B=8,N=4096
    ExpertRouter_gating_kernel<<<nblocks, 256, 0, stream>>>(
        x, gate_w, noise_w, noise_sample, out_w, out_dec, accum);

    ExpertRouter_finalize_kernel<<<1, 1, 0, stream>>>(accum, out_loss, bn);
}
